// Model_47828755808551
// MI455X (gfx1250) — matrix operand order repaired and checked by reading, not compiled, not run on hardware
//
#include <hip/hip_runtime.h>

// ---------------------------------------------------------------------------
// Model dims
// ---------------------------------------------------------------------------
#define BB 16
#define TT 288
#define NN 207
#define PATCH 12
#define PP 24            // T / PATCH
#define SPD 288
#define OUT_STEPS 96
#define EMB 96
#define TODD 24
#define DOWD 24
#define SPA 40
#define POS 24
#define TIME_DIM 72      // TODD+DOWD+POS
#define TGT_DIM 160      // EMB+SPA+POS
#define HH 8
#define FF 1024
#define LL 3
#define MG 79488L        // B*N*P rows of gf
#define MT 384           // B*P rows of tf
#define CHUNK 8192L      // FFN row chunk

typedef __attribute__((ext_vector_type(16))) __bf16 bf16x16;
typedef __attribute__((ext_vector_type(8)))  __bf16 bf16x8;
typedef __attribute__((ext_vector_type(8)))  float  floatx8;

__device__ __forceinline__ __bf16 f2bf(float x) {
  union { float f; unsigned u; } a; a.f = x;
  unsigned r = a.u + 0x7FFFu + ((a.u >> 16) & 1u);   // round-to-nearest-even
  union { unsigned short s; __bf16 b; } o; o.s = (unsigned short)(r >> 16);
  return o.b;
}

__device__ __forceinline__ bf16x16 cat16(bf16x8 lo, bf16x8 hi) {
  return __builtin_shufflevector(lo, hi,
      0, 1, 2, 3, 4, 5, 6, 7, 8, 9, 10, 11, 12, 13, 14, 15);
}

__device__ __forceinline__ floatx8 wmma_bf(bf16x16 a, bf16x16 b, floatx8 acc) {
  return __builtin_amdgcn_wmma_f32_16x16x32_bf16(false, a, false, b,
                                                 (short)0, acc, false, false);
}

// ---------------------------------------------------------------------------
// FAST GEMM: C[M,N] = A_bf16[M,K] @ Wp(bf16 fragment-packed) + bias (+ReLU).
// Requires M%16==0, K%32==0, N%16==0. One wave computes a 16x64 tile
// (4 accumulators). nv==4 path is fully branch-free: 2 b128 A-loads,
// 8 b128 B-loads, 4 WMMAs per k-step; EXEC stays all-ones.
// ---------------------------------------------------------------------------
__global__ void k_gemm_fast(const __bf16* __restrict__ A,
                            const __bf16* __restrict__ Wp,
                            const float* __restrict__ bias,
                            float* __restrict__ C, int ldc,
                            int M, int K, int N, int relu) {
  const int lane = threadIdx.x;
  const int l15  = lane & 15;
  const int hi   = lane >> 4;
  const int tM   = blockIdx.y << 4;
  const int ntn  = N >> 4;
  const int nt0  = blockIdx.x << 2;
  const int nv   = (ntn - nt0 < 4) ? (ntn - nt0) : 4;   // wave-uniform

  const __bf16* ap = A + (size_t)(tM + l15) * K + (hi << 3);
  const __bf16* bp = Wp + (((size_t)nt0 << 5) + lane) * 16;
  const size_t bstep = (size_t)ntn << 9;                // ntn*32*16 per k-step

  floatx8 acc0 = {0.f,0.f,0.f,0.f,0.f,0.f,0.f,0.f};
  floatx8 acc1 = acc0, acc2 = acc0, acc3 = acc0;

  const int ksteps = K >> 5;
  if (nv == 4) {
    // hot path: straight-line, branch-free
    for (int kt = 0; kt < ksteps; ++kt) {
      bf16x8 alo = *(const bf16x8*)(ap);
      bf16x8 ahi = *(const bf16x8*)(ap + 16);
      bf16x8 b0l = *(const bf16x8*)(bp);
      bf16x8 b0h = *(const bf16x8*)(bp + 8);
      bf16x8 b1l = *(const bf16x8*)(bp + 512);
      bf16x8 b1h = *(const bf16x8*)(bp + 520);
      bf16x8 b2l = *(const bf16x8*)(bp + 1024);
      bf16x8 b2h = *(const bf16x8*)(bp + 1032);
      bf16x8 b3l = *(const bf16x8*)(bp + 1536);
      bf16x8 b3h = *(const bf16x8*)(bp + 1544);
      __builtin_prefetch(ap + 128, 0, 1);
      __builtin_prefetch(bp + bstep, 0, 1);
      bf16x16 a = cat16(alo, ahi);
      acc0 = wmma_bf(a, cat16(b0l, b0h), acc0);
      acc1 = wmma_bf(a, cat16(b1l, b1h), acc1);
      acc2 = wmma_bf(a, cat16(b2l, b2h), acc2);
      acc3 = wmma_bf(a, cat16(b3l, b3h), acc3);
      ap += 32;
      bp += bstep;
    }
  } else {
    // ragged right edge (wave-uniform guards)
    for (int kt = 0; kt < ksteps; ++kt) {
      bf16x16 a = cat16(*(const bf16x8*)(ap), *(const bf16x8*)(ap + 16));
      ap += 32;
      acc0 = wmma_bf(a, cat16(*(const bf16x8*)(bp), *(const bf16x8*)(bp + 8)), acc0);
      if (nv > 1)
        acc1 = wmma_bf(a, cat16(*(const bf16x8*)(bp + 512),
                                *(const bf16x8*)(bp + 520)), acc1);
      if (nv > 2)
        acc2 = wmma_bf(a, cat16(*(const bf16x8*)(bp + 1024),
                                *(const bf16x8*)(bp + 1032)), acc2);
      bp += bstep;
    }
  }

  const int rbase = tM + (hi << 3);
#pragma unroll
  for (int j = 0; j < 4; ++j) {
    if (j >= nv) break;                                  // uniform
    floatx8 acc = (j == 0) ? acc0 : (j == 1) ? acc1 : (j == 2) ? acc2 : acc3;
    int col = ((nt0 + j) << 4) + l15;
    float bv = bias[col];
#pragma unroll
    for (int r = 0; r < 8; ++r) {
      float v = acc[r] + bv;
      if (relu) v = fmaxf(v, 0.f);
      C[(size_t)(rbase + r) * ldc + col] = v;
    }
  }
}

// ---------------------------------------------------------------------------
// GUARDED GEMM (tiny shapes only): branch-free via clamped addresses +
// cndmask zeroing. fp32 in, bf16 WMMA compute.
// ---------------------------------------------------------------------------
__global__ void k_gemm_guard(const float* __restrict__ A, int lda,
                             const float* __restrict__ W, int ldw,
                             const float* __restrict__ bias,
                             float* __restrict__ C, int ldc,
                             int M, int K, int N, int relu) {
  const int tN   = blockIdx.x * 16;
  const int tM   = blockIdx.y * 16;
  const int lane = threadIdx.x;
  const int l15  = lane & 15;
  const int hi   = lane >> 4;
  const int arow = tM + l15;
  const int bcol = tN + l15;
  const bool arow_ok = arow < M;
  const bool bcol_ok = bcol < N;
  const float* Arow = A + (size_t)(arow_ok ? arow : (M - 1)) * lda;
  const int bcolc = bcol_ok ? bcol : (N - 1);

  floatx8 acc = {0.f,0.f,0.f,0.f,0.f,0.f,0.f,0.f};
  for (int k0 = 0; k0 < K; k0 += 32) {
    bf16x16 af, bf;
#pragma unroll
    for (int i = 0; i < 16; ++i) {
      int ka = k0 + ((i >> 3) << 4) + (hi << 3) + (i & 7);
      int kac = ka < K ? ka : (K - 1);
      float av = Arow[kac];                              // always in-bounds
      af[i] = f2bf((arow_ok && ka < K) ? av : 0.f);      // v_cndmask, no branch
      int kb = k0 + ((i >> 3) << 4) + (hi << 3) + (i & 7);
      int kbc = kb < K ? kb : (K - 1);
      float bv = W[(size_t)kbc * ldw + bcolc];
      bf[i] = f2bf((bcol_ok && kb < K) ? bv : 0.f);
    }
    acc = __builtin_amdgcn_wmma_f32_16x16x32_bf16(false, af, false, bf,
                                                  (short)0, acc, false, false);
  }
  float bv = bcol_ok ? bias[bcolc] : 0.f;
#pragma unroll
  for (int r = 0; r < 8; ++r) {
    int row = tM + (hi << 3) + r;
    if (row < M && bcol_ok) {
      float v = acc[r] + bv;
      if (relu) v = fmaxf(v, 0.f);
      C[(size_t)row * ldc + bcol] = v;
    }
  }
}

// ---------------------------------------------------------------------------
// Pack W[Ksrc,N] (fp32) into bf16 WMMA B-fragment order, zero-padded to Kpad.
// Packed index i = ((kt*ntn + nt)*32 + lane)*16 + e  ->  k,n source.
// ---------------------------------------------------------------------------
__global__ void k_pack_w(const float* __restrict__ W, __bf16* __restrict__ Wp,
                         int Ksrc, int Kpad, int N) {
  long i = (long)blockIdx.x * blockDim.x + threadIdx.x;
  long total = (long)Kpad * N;
  if (i >= total) return;
  int  e    = (int)(i & 15);
  int  lane = (int)((i >> 4) & 31);
  long t    = i >> 9;
  int  ntn  = N >> 4;
  int  nt   = (int)(t % ntn);
  long kt   = t / ntn;
  long k    = kt * 32 + ((e >> 3) << 4) + ((lane >> 4) << 3) + (e & 7);
  int  n    = (nt << 4) + (lane & 15);
  float v = (k < Ksrc) ? W[k * N + n] : 0.f;
  Wp[i] = f2bf(v);
}

// fp32 -> bf16 streaming convert (total % 4 == 0)
__global__ void k_cvt_bf16(const float* __restrict__ x, __bf16* __restrict__ y,
                           long total) {
  long i = ((long)blockIdx.x * blockDim.x + threadIdx.x) * 4;
  if (i >= total) return;
  float4 v = *(const float4*)(x + i);
  y[i + 0] = f2bf(v.x);
  y[i + 1] = f2bf(v.y);
  y[i + 2] = f2bf(v.z);
  y[i + 3] = f2bf(v.w);
}

// ---------------------------------------------------------------------------
// LayerNorm over D features (row stride ld*), optional residual: out=LN(x+res)
// ---------------------------------------------------------------------------
__global__ void k_layernorm(const float* __restrict__ x, int ldx,
                            const float* __restrict__ res, int ldr,
                            const float* __restrict__ g, const float* __restrict__ b,
                            float* __restrict__ out, int ldo, int M, int D) {
  int row = blockIdx.x;
  if (row >= M) return;
  int lane = threadIdx.x;
  const float* xr = x + (size_t)row * ldx;
  const float* rr = res ? res + (size_t)row * ldr : nullptr;
  float vals[8];
  int cnt = 0;
  float s = 0.f;
  for (int d = lane; d < D; d += 32) {
    float v = xr[d] + (rr ? rr[d] : 0.f);
    vals[cnt++] = v; s += v;
  }
  for (int o = 16; o >= 1; o >>= 1) s += __shfl_xor(s, o, 32);
  float mean = s / (float)D;
  float vs = 0.f;
  cnt = 0;
  for (int d = lane; d < D; d += 32) { float t = vals[cnt++] - mean; vs += t * t; }
  for (int o = 16; o >= 1; o >>= 1) vs += __shfl_xor(vs, o, 32);
  float inv = rsqrtf(vs / (float)D + 1e-5f);
  float* orow = out + (size_t)row * ldo;
  cnt = 0;
  for (int d = lane; d < D; d += 32)
    orow[d] = (vals[cnt++] - mean) * inv * g[d] + b[d];
}

// ---------------------------------------------------------------------------
// Gather patches straight to bf16, zero-padded K 36 -> 64:
// xp[(b,n,p), f] = f<36 ? x[b, p*12+f/3, n, f%3] : 0
// ---------------------------------------------------------------------------
__global__ void k_patch_gather_bf(const float* __restrict__ x,
                                  __bf16* __restrict__ xp) {
  long i = (long)blockIdx.x * blockDim.x + threadIdx.x;
  const long total = MG * 64L;
  if (i >= total) return;
  int  f = (int)(i & 63);
  long r = i >> 6;
  int  p = (int)(r % PP);
  long bn = r / PP;
  int  n = (int)(bn % NN);
  int  b = (int)(bn / NN);
  float v = 0.f;
  if (f < 36) {
    int j = f / 3, c = f % 3;
    int t = p * PATCH + j;
    v = x[(((long)b * TT + t) * NN + n) * 3 + c];
  }
  xp[i] = f2bf(v);
}

// gf[(b,n,p), 96:160] = [node_emb[n], pos_series[p]]
__global__ void k_build_gf_rest(const float* __restrict__ node_emb,
                                const float* __restrict__ pos_series,
                                float* __restrict__ gf) {
  long i = (long)blockIdx.x * blockDim.x + threadIdx.x;
  const long total = MG * 64L;
  if (i >= total) return;
  int  f = (int)(i & 63);
  long r = i >> 6;
  int  p = (int)(r % PP);
  long bn = r / PP;
  int  n = (int)(bn % NN);
  float v = (f < SPA) ? node_emb[n * SPA + f] : pos_series[p * POS + (f - SPA)];
  gf[r * TGT_DIM + EMB + f] = v;
}

// tf[b,p,:] = [tod_emb[idx], dow_emb[idx], pos_time[p]]
__global__ void k_build_tf(const float* __restrict__ x,
                           const float* __restrict__ tod_emb,
                           const float* __restrict__ dow_emb,
                           const float* __restrict__ pos_time,
                           float* __restrict__ tf) {
  int i = blockIdx.x * blockDim.x + threadIdx.x;
  if (i >= BB * PP * TIME_DIM) return;
  int f = i % TIME_DIM;
  int p = (i / TIME_DIM) % PP;
  int b = i / (TIME_DIM * PP);
  float val;
  if (f < TODD) {
    float tval = x[(((long)b * TT + p * PATCH) * NN + 0) * 3 + 1];
    int idx = (int)(tval * (float)SPD);
    idx = idx < 0 ? 0 : (idx > SPD - 1 ? SPD - 1 : idx);
    val = tod_emb[idx * TODD + f];
  } else if (f < TODD + DOWD) {
    float dval = x[(((long)b * TT + p * PATCH) * NN + 0) * 3 + 2];
    int idx = (int)dval;
    idx = idx < 0 ? 0 : (idx > 6 ? 6 : idx);
    val = dow_emb[idx * DOWD + (f - TODD)];
  } else {
    val = pos_time[p * POS + (f - TODD - DOWD)];
  }
  tf[i] = val;
}

// ---------------------------------------------------------------------------
// Time attention, one block per (b,h). head dim 9, seq P=24.
// ---------------------------------------------------------------------------
__global__ void k_time_attn(const float* __restrict__ q, const float* __restrict__ k,
                            const float* __restrict__ v, float* __restrict__ a_t,
                            float* __restrict__ o) {
  __shared__ float ks[PP][9], vs[PP][9];
  int b = blockIdx.x >> 3;
  int h = blockIdx.x & 7;
  int lane = threadIdx.x;
  for (int i = lane; i < PP * 9; i += 32) {
    int p = i / 9, d = i % 9;
    long base = ((long)(b * PP + p)) * TIME_DIM + h * 9 + d;
    ks[p][d] = k[base];
    vs[p][d] = v[base];
  }
  __syncthreads();
  if (lane < PP) {
    int p = lane;
    float qr[9];
    for (int d = 0; d < 9; ++d) qr[d] = q[((long)(b * PP + p)) * TIME_DIM + h * 9 + d];
    float sc[PP];
    float mx = -1e30f;
    for (int s = 0; s < PP; ++s) {
      float t = 0.f;
      for (int d = 0; d < 9; ++d) t += qr[d] * ks[s][d];
      t *= (1.0f / 3.0f);
      sc[s] = t; mx = fmaxf(mx, t);
    }
    float sum = 0.f;
    for (int s = 0; s < PP; ++s) { sc[s] = __expf(sc[s] - mx); sum += sc[s]; }
    float inv = 1.0f / sum;
    long abase = (((long)(b * HH + h)) * PP + p) * PP;
    for (int s = 0; s < PP; ++s) { sc[s] *= inv; a_t[abase + s] = sc[s]; }
    for (int d = 0; d < 9; ++d) {
      float t = 0.f;
      for (int s = 0; s < PP; ++s) t += sc[s] * vs[s][d];
      o[((long)(b * PP + p)) * TIME_DIM + h * 9 + d] = t;
    }
  }
}

// ---------------------------------------------------------------------------
// Graph attention, one block per (b,n,h). head dim 20, seq P=24,
// mixed with a_t: m = w*a_t + (1-w)*softmax(qk).
// ---------------------------------------------------------------------------
__global__ void k_graph_attn(const float* __restrict__ q, const float* __restrict__ k,
                             const float* __restrict__ v, const float* __restrict__ a_t,
                             const float* __restrict__ wp, float* __restrict__ o) {
  __shared__ float ks[PP][20], vs[PP][20];
  int idx = blockIdx.x;
  int h = idx & 7;
  int n = (idx >> 3) % NN;
  int b = idx / (HH * NN);
  int lane = threadIdx.x;
  long rbase = ((long)((b * NN + n) * PP)) * TGT_DIM + h * 20;
  for (int i = lane; i < PP * 20; i += 32) {
    int p = i / 20, d = i % 20;
    ks[p][d] = k[rbase + (long)p * TGT_DIM + d];
    vs[p][d] = v[rbase + (long)p * TGT_DIM + d];
  }
  __syncthreads();
  float w = wp[0];
  if (lane < PP) {
    int p = lane;
    float qr[20];
    for (int d = 0; d < 20; ++d) qr[d] = q[rbase + (long)p * TGT_DIM + d];
    float sc[PP];
    float mx = -1e30f;
    for (int s = 0; s < PP; ++s) {
      float t = 0.f;
      for (int d = 0; d < 20; ++d) t += qr[d] * ks[s][d];
      t *= 0.22360679775f;
      sc[s] = t; mx = fmaxf(mx, t);
    }
    float sum = 0.f;
    for (int s = 0; s < PP; ++s) { sc[s] = __expf(sc[s] - mx); sum += sc[s]; }
    float inv = 1.0f / sum;
    long abase = (((long)(b * HH + h)) * PP + p) * PP;
    for (int s = 0; s < PP; ++s)
      sc[s] = w * a_t[abase + s] + (1.0f - w) * sc[s] * inv;
    for (int d = 0; d < 20; ++d) {
      float t = 0.f;
      for (int s = 0; s < PP; ++s) t += sc[s] * vs[s][d];
      o[rbase + (long)p * TGT_DIM + d] = t;
    }
  }
}

// out[b,s,n] = tmp[b,n,s]
__global__ void k_out_transpose(const float* __restrict__ tmp, float* __restrict__ out) {
  int i = blockIdx.x * blockDim.x + threadIdx.x;
  if (i >= BB * OUT_STEPS * NN) return;
  int n = i % NN;
  int s = (i / NN) % OUT_STEPS;
  int b = i / (NN * OUT_STEPS);
  out[i] = tmp[((long)(b * NN + n)) * OUT_STEPS + s];
}

// ---------------------------------------------------------------------------
// Host orchestration. Input index map assumes JAX pytree flattening
// (dict keys sorted alphabetically at every level): params, x, y on top;
// per-layer 33 leaves in order fc_g(l1 b/w, l2 b/w, ln b/g), fc_t(...),
// g_k(b/w), g_q, g_v, ln1(b/g), ln2, t_k, t_o, t_q, t_v, tp, w.
// ---------------------------------------------------------------------------
extern "C" void kernel_launch(void* const* d_in, const int* in_sizes, int n_in,
                              void* d_out, int out_size, void* d_ws, size_t ws_size,
                              hipStream_t stream) {
  (void)in_sizes; (void)n_in; (void)out_size; (void)ws_size;
  auto F = [&](int i) { return (const float*)d_in[i]; };

  const float* dow_emb    = F(0);
  const float* node_emb   = F(100);
  const float* outp_b     = F(101);
  const float* outp_w     = F(102);
  const float* patch_b    = F(103);
  const float* patch_w    = F(104);
  const float* pln_b      = F(105);
  const float* pln_g      = F(106);
  const float* pos_series = F(107);
  const float* pos_time   = F(108);
  const float* tfc_b      = F(109);
  const float* tfc_w      = F(110);
  const float* tod_emb    = F(111);
  const float* x          = F(112);

  char* base = (char*)d_ws;
  size_t off = 0;
  auto allocB = [&](size_t bytes) {
    off = (off + 255) & ~(size_t)255;
    void* p = base + off; off += bytes; return p;
  };
  auto allocF = [&](size_t n) { return (float*)allocB(n * 4); };
  auto allocH = [&](size_t n) { return (__bf16*)allocB(n * 2); };

  float* gf     = allocF((size_t)MG * TGT_DIM);
  float* bufA   = allocF((size_t)MG * TGT_DIM);   // qg / g_out
  float* bufB   = allocF((size_t)MG * TGT_DIM);   // kg / ffn hidden chunk (f32)
  float* bufC   = allocF((size_t)MG * TGT_DIM);   // vg / ffn out
  float* bufD   = allocF((size_t)MG * TGT_DIM);   // attn output
  float* tf     = allocF(MT * TIME_DIM);
  float* tqb    = allocF(MT * TIME_DIM);
  float* tkb    = allocF(MT * TIME_DIM);
  float* tvb    = allocF(MT * TIME_DIM);
  float* tat    = allocF(MT * TIME_DIM);
  float* ttmp   = allocF(MT * TIME_DIM);
  float* tfh    = allocF(MT * FF);
  float* tffo   = allocF(MT * TIME_DIM);
  float* a_t    = allocF((size_t)BB * HH * PP * PP);
  float* outtmp = allocF((size_t)BB * NN * OUT_STEPS);
  __bf16* abf   = allocH((size_t)MG * TGT_DIM);   // bf16 activations (also patch)
  __bf16* hbf   = allocH((size_t)CHUNK * FF);     // bf16 FFN hidden chunk
  __bf16* wp_q  = allocH(160 * 160);
  __bf16* wp_k  = allocH(160 * 160);
  __bf16* wp_v  = allocH(160 * 160);
  __bf16* wp_tp = allocH(160 * 160);
  __bf16* wp_f1 = allocH(160 * 1024);
  __bf16* wp_f2 = allocH(1024 * 160);
  __bf16* wp_o  = allocH((size_t)3840 * 96);
  __bf16* wp_pa = allocH(64 * 96);

  auto fgemm = [&](const __bf16* A, const __bf16* Wp, const float* bias,
                   float* C, int ldc, int M, int K, int N, int relu) {
    dim3 g(((N >> 4) + 3) / 4, M >> 4);
    k_gemm_fast<<<g, 32, 0, stream>>>(A, Wp, bias, C, ldc, M, K, N, relu);
  };
  auto ggemm = [&](const float* A, int lda, const float* W, int ldw,
                   const float* bias, float* C, int ldc, int M, int K, int N,
                   int relu) {
    dim3 g((N + 15) / 16, (M + 15) / 16);
    k_gemm_guard<<<g, 32, 0, stream>>>(A, lda, W, ldw, bias, C, ldc, M, K, N, relu);
  };
  auto pack = [&](const float* W, __bf16* Wp, int Ksrc, int Kpad, int N) {
    long tot = (long)Kpad * N;
    k_pack_w<<<(int)((tot + 255) / 256), 256, 0, stream>>>(W, Wp, Ksrc, Kpad, N);
  };
  auto cvt = [&](const float* src, __bf16* dst, long total) {
    k_cvt_bf16<<<(int)((total / 4 + 255) / 256), 256, 0, stream>>>(src, dst, total);
  };
  auto lnorm = [&](const float* xx, int ldx, const float* res, int ldr,
                   const float* gg, const float* bb, float* oo, int ldo, int M, int D) {
    k_layernorm<<<M, 32, 0, stream>>>(xx, ldx, res, ldr, gg, bb, oo, ldo, M, D);
  };

  // ---- Embeddings -----------------------------------------------------------
  pack(patch_w, wp_pa, 36, 64, EMB);
  pack(outp_w, wp_o, 3840, 3840, OUT_STEPS);
  {
    long tot = MG * 64L;
    k_patch_gather_bf<<<(int)((tot + 255) / 256), 256, 0, stream>>>(x, abf);
    fgemm(abf, wp_pa, patch_b, gf, TGT_DIM, (int)MG, 64, EMB, 0);
    lnorm(gf, TGT_DIM, nullptr, 0, pln_g, pln_b, gf, TGT_DIM, (int)MG, EMB);
    k_build_gf_rest<<<(int)((tot + 255) / 256), 256, 0, stream>>>(node_emb,
                                                                 pos_series, gf);
    k_build_tf<<<(BB * PP * TIME_DIM + 255) / 256, 256, 0, stream>>>(
        x, tod_emb, dow_emb, pos_time, tf);
  }

  // ---- Layers ---------------------------------------------------------------
  for (int l = 0; l < LL; ++l) {
    int bs = 1 + 33 * l;
    const float *fcg_l1_b = F(bs+0),  *fcg_l1_w = F(bs+1);
    const float *fcg_l2_b = F(bs+2),  *fcg_l2_w = F(bs+3);
    const float *fcg_ln_b = F(bs+4),  *fcg_ln_g = F(bs+5);
    const float *fct_l1_b = F(bs+6),  *fct_l1_w = F(bs+7);
    const float *fct_l2_b = F(bs+8),  *fct_l2_w = F(bs+9);
    const float *fct_ln_b = F(bs+10), *fct_ln_g = F(bs+11);
    const float *gk_b = F(bs+12), *gk_w = F(bs+13);
    const float *gq_b = F(bs+14), *gq_w = F(bs+15);
    const float *gv_b = F(bs+16), *gv_w = F(bs+17);
    const float *ln1_b = F(bs+18), *ln1_g = F(bs+19);
    const float *ln2_b = F(bs+20), *ln2_g = F(bs+21);
    const float *tk_b = F(bs+22), *tk_w = F(bs+23);
    const float *to_b = F(bs+24), *to_w = F(bs+25);
    const float *tq_b = F(bs+26), *tq_w = F(bs+27);
    const float *tv_b = F(bs+28), *tv_w = F(bs+29);
    const float *tp_b = F(bs+30), *tp_w = F(bs+31);
    const float *wmix = F(bs+32);

    // pack this layer's heavy weights
    pack(gq_w, wp_q, 160, 160, 160);
    pack(gk_w, wp_k, 160, 160, 160);
    pack(gv_w, wp_v, 160, 160, 160);
    pack(tp_w, wp_tp, 160, 160, 160);
    pack(fcg_l1_w, wp_f1, 160, 160, FF);
    pack(fcg_l2_w, wp_f2, FF, FF, 160);

    // time side (produces a_t used by the graph side) -- tiny, guarded path
    ggemm(tf, TIME_DIM, tq_w, TIME_DIM, tq_b, tqb, TIME_DIM, MT, TIME_DIM, TIME_DIM, 0);
    ggemm(tf, TIME_DIM, tk_w, TIME_DIM, tk_b, tkb, TIME_DIM, MT, TIME_DIM, TIME_DIM, 0);
    ggemm(tf, TIME_DIM, tv_w, TIME_DIM, tv_b, tvb, TIME_DIM, MT, TIME_DIM, TIME_DIM, 0);
    k_time_attn<<<BB * HH, 32, 0, stream>>>(tqb, tkb, tvb, a_t, tat);
    ggemm(tat, TIME_DIM, to_w, TIME_DIM, to_b, ttmp, TIME_DIM, MT, TIME_DIM, TIME_DIM, 0);
    lnorm(ttmp, TIME_DIM, tf, TIME_DIM, ln1_g, ln1_b, tf, TIME_DIM, MT, TIME_DIM);
    ggemm(tf, TIME_DIM, fct_l1_w, FF, fct_l1_b, tfh, FF, MT, TIME_DIM, FF, 1);
    ggemm(tfh, FF, fct_l2_w, TIME_DIM, fct_l2_b, tffo, TIME_DIM, MT, FF, TIME_DIM, 0);
    lnorm(tffo, TIME_DIM, tf, TIME_DIM, fct_ln_g, fct_ln_b, tf, TIME_DIM, MT, TIME_DIM);

    // graph side -- fast WMMA path
    cvt(gf, abf, MG * TGT_DIM);
    fgemm(abf, wp_q, gq_b, bufA, TGT_DIM, (int)MG, TGT_DIM, TGT_DIM, 0);
    fgemm(abf, wp_k, gk_b, bufB, TGT_DIM, (int)MG, TGT_DIM, TGT_DIM, 0);
    fgemm(abf, wp_v, gv_b, bufC, TGT_DIM, (int)MG, TGT_DIM, TGT_DIM, 0);
    k_graph_attn<<<BB * NN * HH, 32, 0, stream>>>(bufA, bufB, bufC, a_t, wmix, bufD);
    cvt(bufD, abf, MG * TGT_DIM);
    fgemm(abf, wp_tp, tp_b, bufA, TGT_DIM, (int)MG, TGT_DIM, TGT_DIM, 0);
    lnorm(bufA, TGT_DIM, gf, TGT_DIM, ln2_g, ln2_b, gf, TGT_DIM, (int)MG, TGT_DIM);

    // graph FFN, chunked
    cvt(gf, abf, MG * TGT_DIM);
    for (long c0 = 0; c0 < MG; c0 += CHUNK) {
      int rows = (int)((MG - c0 < CHUNK) ? (MG - c0) : CHUNK);
      fgemm(abf + c0 * TGT_DIM, wp_f1, fcg_l1_b, bufB, FF, rows, TGT_DIM, FF, 1);
      cvt(bufB, hbf, (long)rows * FF);
      fgemm(hbf, wp_f2, fcg_l2_b, bufC + c0 * TGT_DIM, TGT_DIM, rows, FF, TGT_DIM, 0);
    }
    lnorm(bufC, TGT_DIM, gf, TGT_DIM, fcg_ln_g, fcg_ln_b, gf, TGT_DIM, (int)MG, TGT_DIM);
  }

  // ---- Heads ----------------------------------------------------------------
  // out = gf.reshape(B*N, P*160) @ out_proj + b  (abf reinterpreted [B*N, 3840])
  cvt(gf, abf, MG * TGT_DIM);
  fgemm(abf, wp_o, outp_b, outtmp, OUT_STEPS, BB * NN, PP * TGT_DIM, OUT_STEPS, 0);
  k_out_transpose<<<(BB * OUT_STEPS * NN + 255) / 256, 256, 0, stream>>>(
      outtmp, (float*)d_out);
  // time = tf.reshape(B, P*72) @ time_fc + b -> (B, 96, 2)
  ggemm(tf, PP * TIME_DIM, tfc_w, OUT_STEPS * 2, tfc_b,
        (float*)d_out + (long)BB * OUT_STEPS * NN, OUT_STEPS * 2,
        BB, PP * TIME_DIM, OUT_STEPS * 2, 0);
}
